// Model_16801912062284
// MI455X (gfx1250) — matrix operand order repaired and checked by reading, not compiled, not run on hardware
//
#include <hip/hip_runtime.h>
#include <hip/hip_bf16.h>

typedef __attribute__((ext_vector_type(16))) __bf16    v16bf;
typedef __attribute__((ext_vector_type(8)))  float     v8f;
typedef __attribute__((ext_vector_type(8)))  unsigned  v8u;
typedef __attribute__((ext_vector_type(4)))  unsigned  u32x4;
typedef __attribute__((ext_vector_type(8)))  int       i32x8;
typedef __attribute__((ext_vector_type(4)))  int       i32x4;

#define D512 512
#define TLEN 512
#define HSTR 520   // padded LDS row stride (halfs) to spread banks

#if __has_builtin(__builtin_amdgcn_tensor_load_to_lds) && __has_builtin(__builtin_amdgcn_s_wait_tensorcnt)
#define USE_TDM 1
#endif

static __device__ __forceinline__ v8f wmma_bf16(v16bf a, v16bf b, v8f c) {
    return __builtin_amdgcn_wmma_f32_16x16x32_bf16(
        /*neg_a=*/false, a, /*neg_b=*/false, b,
        /*c_mod=*/(short)0, c, /*reuse_a=*/false, /*reuse_b=*/false);
}

static __device__ __forceinline__ unsigned pack2_bf16(float a, float b) {
    union { __bf16 h[2]; unsigned u; } cu;
    cu.h[0] = (__bf16)a; cu.h[1] = (__bf16)b;
    return cu.u;
}
static __device__ __forceinline__ float bf16u_to_f(unsigned short s) {
    union { unsigned u; float f; } c; c.u = ((unsigned)s) << 16; return c.f;
}

#ifdef USE_TDM
// ---------------------------------------------------------------------------
// Issue one TDM 2-D tile load: 16 rows x 512 f32, row stride T*D elements,
// global -> LDS (lds_off = byte offset in LDS). Descriptor per ISA ch.8.
// ---------------------------------------------------------------------------
static __device__ __forceinline__ void tdm_load_tile(const float* gsrc, unsigned lds_off)
{
    unsigned long long ga = (unsigned long long)(uintptr_t)gsrc;
    u32x4 g0;
    g0[0] = 1u;                                                // count=1 (valid), user mode
    g0[1] = lds_off;                                           // LDS byte address
    g0[2] = (unsigned)(ga & 0xFFFFFFFFull);                    // global_addr[31:0]
    g0[3] = (unsigned)((ga >> 32) & 0x1FFFFFFull) | 0x80000000u; // addr[56:32] | type=2
    i32x8 g1;
    g1[0] = 0x00020000;            // workgroup_mask=0, data_size=2 (4 bytes)
    g1[1] = (int)(512u << 16);     // tensor_dim0 = 512 (bits 63:48)
    g1[2] = (int)(16u  << 16);     // tensor_dim1 = 16  (bits 95:80)
    g1[3] = (int)(512u << 16);     // tile_dim0   = 512 (bits 127:112)
    g1[4] = 16;                    // tile_dim1 = 16, tile_dim2 = 0
    g1[5] = (int)262144;           // tensor_dim0_stride = T*D (elements)
    g1[6] = 0;                     // stride0 hi / stride1 lo
    g1[7] = 0;                     // stride1 hi
    i32x4 gz = {0, 0, 0, 0};       // 2-D tensor: groups 2/3 unused
#if __clang_major__ >= 23
    i32x8 gz8 = {0, 0, 0, 0, 0, 0, 0, 0};
    __builtin_amdgcn_tensor_load_to_lds(g0, g1, gz, gz, gz8, 0);
#else
    __builtin_amdgcn_tensor_load_to_lds(g0, g1, gz, gz, 0);
#endif
}
#endif

// ---------------------------------------------------------------------------
// Pack a [512, ldw] fp32 weight slice (cols col_off..col_off+511) into the
// CDNA5 WMMA B-operand bf16 layout:
//   dword index = ((nt*16 + kt)*32 + lane)*8 + v
//   n = nt*16 + (lane&15);  k = kt*32 + (v<4 ? 2v : 16+2(v-4)) + (lane>>4)*8 (+1 in high half)
// ---------------------------------------------------------------------------
__global__ __launch_bounds__(256) void pack_b_kernel(
    const float* __restrict__ W, int ldw, int col_off, unsigned* __restrict__ out)
{
    int idx  = blockIdx.x * 256 + threadIdx.x;      // 0 .. 131071
    int v    = idx & 7;
    int lane = (idx >> 3) & 31;
    int kt   = (idx >> 8) & 15;
    int nt   = idx >> 12;
    int n    = nt * 16 + (lane & 15);
    int k0   = kt * 32 + ((v < 4) ? v * 2 : 16 + (v - 4) * 2) + (lane >> 4) * 8;
    const float* row = W + (long)n * ldw + col_off;
    out[idx] = pack2_bf16(row[k0], row[k0 + 1]);
}

// ---------------------------------------------------------------------------
// Layer-0 X-projection (K = 21, trivial): Xg/Xa = x_enc @ Wx.T + b
// ---------------------------------------------------------------------------
__global__ __launch_bounds__(256) void xproj0_kernel(
    const float* __restrict__ x, const float* __restrict__ Wg,
    const float* __restrict__ Wa, const float* __restrict__ bg,
    const float* __restrict__ ba, float* __restrict__ Xg, float* __restrict__ Xa)
{
    int i = blockIdx.x >> 1;                               // 0 .. B*T-1
    int d = ((blockIdx.x & 1) << 8) + threadIdx.x;         // 0 .. 511
    float sg = bg[d], sa = ba[d];
    const float* xr = x + (long)i * 21;
    const float* wg = Wg + (long)d * 533 + 512;
    const float* wa = Wa + (long)d * 533 + 512;
    #pragma unroll
    for (int k = 0; k < 21; ++k) {
        float xv = xr[k];
        sg += xv * wg[k];
        sa += xv * wa[k];
    }
    Xg[(long)i * D512 + d] = sg;
    Xa[(long)i * D512 + d] = sa;
}

// ---------------------------------------------------------------------------
// Layers 1/2 X-projection: big WMMA GEMM  [32768,512] x [512,512] for 2 gates
// One wave = one 16x16 output tile for both gates.
// ---------------------------------------------------------------------------
__global__ __launch_bounds__(256) void xproj_big_kernel(
    const unsigned short* __restrict__ ybf,
    const unsigned* __restrict__ Wxg_pk, const unsigned* __restrict__ Wxa_pk,
    const float* __restrict__ bg, const float* __restrict__ ba,
    float* __restrict__ Xg, float* __restrict__ Xa)
{
    const int lane  = threadIdx.x & 31;
    const int wv    = threadIdx.x >> 5;
    const int gw    = blockIdx.x * 8 + wv;   // 0 .. 65535
    const int mtile = gw >> 5;               // 0 .. 2047
    const int nt    = gw & 31;               // 0 .. 31
    const int khalf = lane >> 4;
    const int cc    = lane & 15;
    const unsigned* yu = (const unsigned*)ybf;
    const int arow = mtile * 16 + cc;

    v8f accG = {}, accA = {};
    #pragma unroll 4
    for (int kt = 0; kt < 16; ++kt) {
        const int ktb = kt * 32;
        v8u au;
        #pragma unroll
        for (int v = 0; v < 8; ++v) {
            int k0 = ktb + ((v < 4) ? v * 2 : 16 + (v - 4) * 2) + khalf * 8;
            au[v] = yu[((long)arow * D512 + k0) >> 1];
        }
        v16bf a  = __builtin_bit_cast(v16bf, au);
        v16bf Bg = __builtin_bit_cast(v16bf, *(const v8u*)(Wxg_pk + (((nt * 16 + kt) * 32 + lane) << 3)));
        v16bf Ba = __builtin_bit_cast(v16bf, *(const v8u*)(Wxa_pk + (((nt * 16 + kt) * 32 + lane) << 3)));
        accG = wmma_bf16(a, Bg, accG);
        accA = wmma_bf16(a, Ba, accA);
    }
    const int col = nt * 16 + cc;
    const float bgv = bg[col], bav = ba[col];
    #pragma unroll
    for (int r = 0; r < 8; ++r) {
        long row = (long)(mtile * 16 + r + 8 * khalf);
        Xg[row * D512 + col] = accG[r] + bgv;
        Xa[row * D512 + col] = accA[r] + bav;
    }
}

// ---------------------------------------------------------------------------
// Sequential recurrence: one workgroup (16 waves) owns 16 batch rows for all
// 512 timesteps. h lives in LDS (bf16); weights stream from L2 in packed
// B-operand layout. X-streams are staged into LDS one step ahead by the
// Tensor Data Mover (double-buffered), overlapping DMA with the WMMA chain.
// ---------------------------------------------------------------------------
__global__ __launch_bounds__(512) void recurrence_kernel(
    const unsigned* __restrict__ Wg_pk, const unsigned* __restrict__ Wa_pk,
    const float* __restrict__ Xg, const float* __restrict__ Xa,
    const float* __restrict__ ln_g, const float* __restrict__ ln_b,
    unsigned short* __restrict__ y_bf16)
{
    __shared__ unsigned short h_lds[16 * HSTR];
    __shared__ float red_s[16 * 16];
    __shared__ float red_q[16 * 16];
    __shared__ float stat_mu[16];
    __shared__ float stat_rs[16];
#ifdef USE_TDM
    extern __shared__ float xdyn[];   // 2 stages x (Xg 16x512 + Xa 16x512) = 128 KiB
#endif

    const int tid   = threadIdx.x;
    const int w     = tid >> 5;          // wave 0..15
    const int lane  = tid & 31;
    const int khalf = lane >> 4;
    const int c     = lane & 15;
    const int b0    = blockIdx.x * 16;   // batch-row base
    const int n0    = w * 2, n1 = w * 2 + 1;
    const int col0  = n0 * 16 + c, col1 = n1 * 16 + c;

    for (int i = tid; i < 16 * HSTR; i += 512) h_lds[i] = 0;  // h0 = 0
    __syncthreads();

    const unsigned* hlds_u = (const unsigned*)h_lds;
    const float lg0 = ln_g[col0], lb0 = ln_b[col0];
    const float lg1 = ln_g[col1], lb1 = ln_b[col1];

#ifdef USE_TDM
    const long xbase = (long)b0 * TLEN * D512;   // row stride handled by descriptor
    if (w == 0) {   // prologue: stage 0 <- step 0
        tdm_load_tile(Xg + xbase, (unsigned)(uintptr_t)&xdyn[0]);
        tdm_load_tile(Xa + xbase, (unsigned)(uintptr_t)&xdyn[8192]);
    }
#endif

    for (int t = 0; t < TLEN; ++t) {
#ifdef USE_TDM
        // kick off next step's X DMA into the other stage (overlaps WMMA)
        if (w == 0 && (t + 1) < TLEN) {
            const unsigned stg = ((t + 1) & 1) * 16384u;
            tdm_load_tile(Xg + xbase + (long)(t + 1) * D512,
                          (unsigned)(uintptr_t)&xdyn[stg]);
            tdm_load_tile(Xa + xbase + (long)(t + 1) * D512,
                          (unsigned)(uintptr_t)&xdyn[stg + 8192u]);
        }
#endif
        // --- phase 1: h @ Wh.T for both gates, 4 tiles per wave -------------
        v8f accG0 = {}, accG1 = {}, accA0 = {}, accA1 = {};
        #pragma unroll 4
        for (int kt = 0; kt < 16; ++kt) {
            const int ktb = kt * 32;
            v8u au;
            #pragma unroll
            for (int v = 0; v < 8; ++v) {
                int k0 = ktb + ((v < 4) ? v * 2 : 16 + (v - 4) * 2) + khalf * 8;
                au[v] = hlds_u[(c * HSTR + k0) >> 1];
            }
            v16bf a = __builtin_bit_cast(v16bf, au);
            v16bf Bg0 = __builtin_bit_cast(v16bf, *(const v8u*)(Wg_pk + (((n0 * 16 + kt) * 32 + lane) << 3)));
            v16bf Bg1 = __builtin_bit_cast(v16bf, *(const v8u*)(Wg_pk + (((n1 * 16 + kt) * 32 + lane) << 3)));
            v16bf Ba0 = __builtin_bit_cast(v16bf, *(const v8u*)(Wa_pk + (((n0 * 16 + kt) * 32 + lane) << 3)));
            v16bf Ba1 = __builtin_bit_cast(v16bf, *(const v8u*)(Wa_pk + (((n1 * 16 + kt) * 32 + lane) << 3)));
            accG0 = wmma_bf16(a, Bg0, accG0);
            accG1 = wmma_bf16(a, Bg1, accG1);
            accA0 = wmma_bf16(a, Ba0, accA0);
            accA1 = wmma_bf16(a, Ba1, accA1);
        }

#ifdef USE_TDM
        // current step's tiles must have landed (in-order TENSORcnt: leave the
        // two next-step DMAs in flight, wait for the rest)
        if (w == 0) {
            if ((t + 1) < TLEN) __builtin_amdgcn_s_wait_tensorcnt(2);
            else                __builtin_amdgcn_s_wait_tensorcnt(0);
        }
        __syncthreads();
        const float* xs = xdyn + (unsigned)(t & 1) * 16384u;   // [Xg 16x512 | Xa 16x512]
#else
        if (t + 1 < TLEN) {
            long gpre = ((long)(b0 + khalf * 8) * TLEN + (t + 1)) * D512 + col0;
            __builtin_prefetch(Xg + gpre, 0, 1);
            __builtin_prefetch(Xa + gpre, 0, 1);
        }
#endif

        // --- phase 2: add precomputed x-part, gate + mix --------------------
        float ht0[8], ht1[8], s8[8], q8[8];
        #pragma unroll
        for (int r = 0; r < 8; ++r) {
            const int row = r + 8 * khalf;
#ifdef USE_TDM
            float g0 = accG0[r] + xs[row * D512 + col0];
            float a0 = accA0[r] + xs[8192 + row * D512 + col0];
            float g1 = accG1[r] + xs[row * D512 + col1];
            float a1 = accA1[r] + xs[8192 + row * D512 + col1];
#else
            const long gi = ((long)(b0 + row) * TLEN + t) * D512;
            float g0 = accG0[r] + Xg[gi + col0];
            float a0 = accA0[r] + Xa[gi + col0];
            float g1 = accG1[r] + Xg[gi + col1];
            float a1 = accA1[r] + Xa[gi + col1];
#endif
            float al0 = 1.f / (1.f + __expf(-g0));
            float h0 = al0 * tanhf(a0) + (1.f - al0) * a0;
            float al1 = 1.f / (1.f + __expf(-g1));
            float h1 = al1 * tanhf(a1) + (1.f - al1) * a1;
            ht0[r] = h0; ht1[r] = h1;
            s8[r] = h0 + h1;
            q8[r] = h0 * h0 + h1 * h1;
        }

        // --- phase 3: LayerNorm stats (intra-wave shfl + LDS across waves) --
        #pragma unroll
        for (int r = 0; r < 8; ++r) {
            #pragma unroll
            for (int off = 1; off < 16; off <<= 1) {
                s8[r] += __shfl_xor(s8[r], off, 32);
                q8[r] += __shfl_xor(q8[r], off, 32);
            }
        }
        if (c == 0) {
            #pragma unroll
            for (int r = 0; r < 8; ++r) {
                const int row = r + 8 * khalf;
                red_s[row * 16 + w] = s8[r];
                red_q[row * 16 + w] = q8[r];
            }
        }
        __syncthreads();
        if (tid < 16) {
            float S = 0.f, Q = 0.f;
            #pragma unroll
            for (int i = 0; i < 16; ++i) { S += red_s[tid * 16 + i]; Q += red_q[tid * 16 + i]; }
            float mu  = S * (1.f / 512.f);
            float var = Q * (1.f / 512.f) - mu * mu;
            stat_mu[tid] = mu;
            stat_rs[tid] = rsqrtf(var + 1e-5f);
        }
        __syncthreads();

        // --- phase 4: normalize, write h (LDS, bf16) + layer output ---------
        #pragma unroll
        for (int r = 0; r < 8; ++r) {
            const int row = r + 8 * khalf;
            const float mu = stat_mu[row], rs = stat_rs[row];
            float hn0 = (ht0[r] - mu) * rs * lg0 + lb0;
            float hn1 = (ht1[r] - mu) * rs * lg1 + lb1;
            union { __bf16 h; unsigned short u; } cu0, cu1;
            cu0.h = (__bf16)hn0; cu1.h = (__bf16)hn1;
            h_lds[row * HSTR + col0] = cu0.u;
            h_lds[row * HSTR + col1] = cu1.u;
            const long gi = ((long)(b0 + row) * TLEN + t) * D512;
            y_bf16[gi + col0] = cu0.u;
            y_bf16[gi + col1] = cu1.u;
        }
        __syncthreads();
    }
}

// ---------------------------------------------------------------------------
// Final projection: out[b, tt, c] = y[b, 416+tt, :] @ W_proj[:, c] + b_proj
// ---------------------------------------------------------------------------
__global__ __launch_bounds__(256) void proj_kernel(
    const unsigned short* __restrict__ ybf, const float* __restrict__ Wp,
    const float* __restrict__ bp, float* __restrict__ out)
{
    int idx = blockIdx.x * 256 + threadIdx.x;
    if (idx >= 64 * 96 * 21) return;
    int cch = idx % 21;
    int tt  = (idx / 21) % 96;
    int b   = idx / (21 * 96);
    int t   = 416 + tt;
    const unsigned short* yr = ybf + ((long)(b * TLEN + t)) * D512;
    float s = bp[cch];
    for (int d = 0; d < D512; ++d)
        s += bf16u_to_f(yr[d]) * Wp[d * 21 + cch];
    out[idx] = s;
}

// ---------------------------------------------------------------------------
extern "C" void kernel_launch(void* const* d_in, const int* in_sizes, int n_in,
                              void* d_out, int out_size, void* d_ws, size_t ws_size,
                              hipStream_t stream)
{
    const float* x_enc = (const float*)d_in[0];
    const float* W_g0 = (const float*)d_in[1];  const float* W_a0 = (const float*)d_in[2];
    const float* b_g0 = (const float*)d_in[3];  const float* b_a0 = (const float*)d_in[4];
    const float* lng0 = (const float*)d_in[5];  const float* lnb0 = (const float*)d_in[6];
    const float* W_g1 = (const float*)d_in[7];  const float* W_a1 = (const float*)d_in[8];
    const float* b_g1 = (const float*)d_in[9];  const float* b_a1 = (const float*)d_in[10];
    const float* lng1 = (const float*)d_in[11]; const float* lnb1 = (const float*)d_in[12];
    const float* W_g2 = (const float*)d_in[13]; const float* W_a2 = (const float*)d_in[14];
    const float* b_g2 = (const float*)d_in[15]; const float* b_a2 = (const float*)d_in[16];
    const float* lng2 = (const float*)d_in[17]; const float* lnb2 = (const float*)d_in[18];
    const float* W_pr = (const float*)d_in[19]; const float* b_pr = (const float*)d_in[20];

    char* ws = (char*)d_ws;
    float*          Xg  = (float*)ws;                               // 64 MiB
    float*          Xa  = (float*)(ws + 67108864ull);               // 64 MiB
    unsigned short* ybf = (unsigned short*)(ws + 134217728ull);     // 32 MiB
    unsigned*       pk  = (unsigned*)(ws + 167772160ull);           // 10 x 512 KiB
    const int PKS = 131072;   // dwords per packed matrix
    unsigned *Whg0 = pk,           *Wha0 = pk + 1 * PKS;
    unsigned *Whg1 = pk + 2 * PKS, *Wha1 = pk + 3 * PKS;
    unsigned *Whg2 = pk + 4 * PKS, *Wha2 = pk + 5 * PKS;
    unsigned *Wxg1 = pk + 6 * PKS, *Wxa1 = pk + 7 * PKS;
    unsigned *Wxg2 = pk + 8 * PKS, *Wxa2 = pk + 9 * PKS;

    // Pack hidden-state & input-projection weights into WMMA B-operand bf16.
    pack_b_kernel<<<512, 256, 0, stream>>>(W_g0,  533,   0, Whg0);
    pack_b_kernel<<<512, 256, 0, stream>>>(W_a0,  533,   0, Wha0);
    pack_b_kernel<<<512, 256, 0, stream>>>(W_g1, 1024,   0, Whg1);
    pack_b_kernel<<<512, 256, 0, stream>>>(W_a1, 1024,   0, Wha1);
    pack_b_kernel<<<512, 256, 0, stream>>>(W_g2, 1024,   0, Whg2);
    pack_b_kernel<<<512, 256, 0, stream>>>(W_a2, 1024,   0, Wha2);
    pack_b_kernel<<<512, 256, 0, stream>>>(W_g1, 1024, 512, Wxg1);
    pack_b_kernel<<<512, 256, 0, stream>>>(W_a1, 1024, 512, Wxa1);
    pack_b_kernel<<<512, 256, 0, stream>>>(W_g2, 1024, 512, Wxg2);
    pack_b_kernel<<<512, 256, 0, stream>>>(W_a2, 1024, 512, Wxa2);

    const size_t dynLds = 131072;   // 2-stage X staging buffers (TDM path)

    // Layer 0
    xproj0_kernel<<<65536, 256, 0, stream>>>(x_enc, W_g0, W_a0, b_g0, b_a0, Xg, Xa);
    recurrence_kernel<<<4, 512, dynLds, stream>>>(Whg0, Wha0, Xg, Xa, lng0, lnb0, ybf);
    // Layer 1
    xproj_big_kernel<<<8192, 256, 0, stream>>>(ybf, Wxg1, Wxa1, b_g1, b_a1, Xg, Xa);
    recurrence_kernel<<<4, 512, dynLds, stream>>>(Whg1, Wha1, Xg, Xa, lng1, lnb1, ybf);
    // Layer 2
    xproj_big_kernel<<<8192, 256, 0, stream>>>(ybf, Wxg2, Wxa2, b_g2, b_a2, Xg, Xa);
    recurrence_kernel<<<4, 512, dynLds, stream>>>(Whg2, Wha2, Xg, Xa, lng2, lnb2, ybf);
    // Output projection (last 96 timesteps)
    proj_kernel<<<504, 256, 0, stream>>>(ybf, W_pr, b_pr, (float*)d_out);
}
